// SelfAttention_8332236554347
// MI455X (gfx1250) — matrix operand order repaired and checked by reading, not compiled, not run on hardware
//
#include <hip/hip_runtime.h>
#include <hip/hip_bf16.h>

#define DIM   512
#define HEADS 8
#define HD    64
#define BATCH 4
#define SEQ   4096   // 64*64

typedef __attribute__((ext_vector_type(16))) __bf16       v16bf;
typedef __attribute__((ext_vector_type(8)))  float        v8f;
typedef __attribute__((ext_vector_type(4)))  unsigned int v4u;

union Frag16 {
    v16bf v;
    struct { v4u lo; v4u hi; } u;
};

// Two 16-byte chunks per lane (B-fragments: K at {p, p+16} halves of 8).
__device__ inline v16bf frag_contig(const __bf16* p) {
    Frag16 f;
    f.u.lo = *reinterpret_cast<const v4u*>(p);
    f.u.hi = *reinterpret_cast<const v4u*>(p + 16);
    return f.v;
}
// Two 16-byte chunks (A-fragments: K split into {base, base+16} halves of 8).
__device__ inline v16bf frag_split(const __bf16* p0, const __bf16* p1) {
    Frag16 f;
    f.u.lo = *reinterpret_cast<const v4u*>(p0);
    f.u.hi = *reinterpret_cast<const v4u*>(p1);
    return f.v;
}

__device__ inline v8f wmma_bf16(v16bf a, v16bf b, v8f c) {
    return __builtin_amdgcn_wmma_f32_16x16x32_bf16(
        false, a, false, b, (short)0, c, false, false);
}

// ---------------------------------------------------------------------------
// Prep kernels
// ---------------------------------------------------------------------------
__global__ void convert_f32_bf16(const float* __restrict__ in,
                                 __bf16* __restrict__ out, int n) {
    int i = blockIdx.x * 256 + threadIdx.x;
    if (i < n) out[i] = (__bf16)in[i];
}

// x[b,c,n] f32 -> xT[b,n,c] bf16
__global__ void transpose_x(const float* __restrict__ x, __bf16* __restrict__ xT) {
    long i = (long)blockIdx.x * 256 + threadIdx.x;
    long total = (long)BATCH * SEQ * DIM;
    if (i < total) {
        int c = (int)(i % DIM);
        long t = i / DIM;
        int n = (int)(t % SEQ);
        int b = (int)(t / SEQ);
        xT[i] = (__bf16)x[((long)b * DIM + c) * SEQ + n];
    }
}

// ---------------------------------------------------------------------------
// QKV GEMM: qkv[n, m] = sum_c xT[n,c] * wqkv[m,c] + bias[m]
// Q scaled by hd^-0.5; V stored transposed [b,h,d,n].
// Block = 4 waves; wave handles 16 rows x 64 cols.
// ---------------------------------------------------------------------------
__global__ void qkv_gemm(const __bf16* __restrict__ xT, const __bf16* __restrict__ wqkv,
                         const float* __restrict__ bqkv,
                         __bf16* __restrict__ Q, __bf16* __restrict__ K,
                         __bf16* __restrict__ Vt) {
    const int lane = threadIdx.x & 31;
    const int wave = threadIdx.x >> 5;
    int bid = blockIdx.x;
    const int colTiles = (3 * DIM) / 256;          // 6
    int cb = bid % colTiles;  bid /= colTiles;
    int n0 = (bid % (SEQ / 16)) * 16;  bid /= (SEQ / 16);
    int b = bid;
    int m0 = cb * 256 + wave * 64;

    const int arow = lane & 15;
    const int asel = (lane >> 4) * 8;
    const int bcol = lane & 15;
    const int bk   = (lane >> 4) * 8;
    const __bf16* aptr = xT + ((long)b * SEQ + n0 + arow) * DIM;

    v8f acc[4] = {};
    for (int k0 = 0; k0 < DIM; k0 += 32) {
        v16bf af = frag_split(aptr + k0 + asel, aptr + k0 + 16 + asel);
#pragma unroll
        for (int t = 0; t < 4; ++t) {
            const __bf16* bptr = wqkv + (long)(m0 + t * 16 + bcol) * DIM + k0 + bk;
            acc[t] = wmma_bf16(af, frag_contig(bptr), acc[t]);
        }
    }

    const int rbase = (lane >> 4) * 8;
#pragma unroll
    for (int t = 0; t < 4; ++t) {
        int mcol  = m0 + t * 16 + bcol;
        float bias = bqkv[mcol];
        int which = mcol / DIM;        // 0=q, 1=k, 2=v  (uniform per tile)
        int mloc  = mcol % DIM;
        int h = mloc / HD, d = mloc % HD;
#pragma unroll
        for (int r = 0; r < 8; ++r) {
            int n = n0 + rbase + r;
            float val = acc[t][r] + bias;
            if (which == 0) {
                Q[(((long)b * HEADS + h) * SEQ + n) * HD + d] = (__bf16)(val * 0.125f);
            } else if (which == 1) {
                K[(((long)b * HEADS + h) * SEQ + n) * HD + d] = (__bf16)val;
            } else {
                Vt[(((long)b * HEADS + h) * HD + d) * SEQ + n] = (__bf16)val;
            }
        }
    }
}

// ---------------------------------------------------------------------------
// Flash attention, ONE-PASS softmax.
// Reference clips S to [-10,10] before softmax => exp(s) in [4.5e-5, 2.2e4],
// row sums <= ~9e7: no max subtraction needed, no O-rescale, no per-iteration
// cross-lane reductions. Each lane keeps private partial row sums; a single
// shuffle reduction at the epilogue produces the normalizers.
// 1 wave = 16 query rows; block = 4 waves; grid = B*H*(N/64).
// ---------------------------------------------------------------------------
__global__ void flash_attn(const __bf16* __restrict__ Q, const __bf16* __restrict__ K,
                           const __bf16* __restrict__ Vt, __bf16* __restrict__ Oa) {
    __shared__ __align__(32) __bf16 Pl[4][16][32];  // per-wave P staging (C->A layout)

    const int lane = threadIdx.x & 31;
    const int wave = threadIdx.x >> 5;
    int bid = blockIdx.x;
    int nt = bid % (SEQ / 64);  bid /= (SEQ / 64);
    int h  = bid % HEADS;       bid /= HEADS;
    int b  = bid;
    int n0 = nt * 64 + wave * 16;

    const __bf16* Qb = Q  + ((long)b * HEADS + h) * SEQ * HD;
    const __bf16* Kb = K  + ((long)b * HEADS + h) * SEQ * HD;
    const __bf16* Vb = Vt + ((long)b * HEADS + h) * (long)HD * SEQ;

    const int arow = lane & 15;
    const int asel = (lane >> 4) * 8;
    const int bcol = lane & 15;
    const int bk   = (lane >> 4) * 8;

    // Q tile 16x64 as two A-fragments (K = 0..31, 32..63); Q pre-scaled.
    const __bf16* qrow = Qb + (long)(n0 + arow) * HD;
    v16bf qf0 = frag_split(qrow + asel,      qrow + 16 + asel);
    v16bf qf1 = frag_split(qrow + 32 + asel, qrow + 48 + asel);

    v8f accO[4] = {};
    float lsum[8];
#pragma unroll
    for (int r = 0; r < 8; ++r) lsum[r] = 0.0f;

    const int rowoff = (lane >> 4) * 8;
    const __bf16* prow = &Pl[wave][arow][0];

    for (int m0 = 0; m0 < SEQ; m0 += 32) {
        // S = Q Kt : two 16x16 f32 tiles (key cols m0..15 and m0+16..31)
        v8f s0 = {}, s1 = {};
        {
            const __bf16* kr0 = Kb + (long)(m0 + bcol) * HD + bk;
            const __bf16* kr1 = Kb + (long)(m0 + 16 + bcol) * HD + bk;
            s0 = wmma_bf16(qf0, frag_contig(kr0), s0);
            s0 = wmma_bf16(qf1, frag_contig(kr0 + 32), s0);
            s1 = wmma_bf16(qf0, frag_contig(kr1), s1);
            s1 = wmma_bf16(qf1, frag_contig(kr1 + 32), s1);
        }

        // P = exp(clip(S)); accumulate private partial row sums; stage to LDS.
#pragma unroll
        for (int r = 0; r < 8; ++r) {
            float p0 = __expf(fminf(fmaxf(s0[r], -10.0f), 10.0f));
            float p1 = __expf(fminf(fmaxf(s1[r], -10.0f), 10.0f));
            lsum[r] += p0 + p1;
            Pl[wave][rowoff + r][bcol]      = (__bf16)p0;
            Pl[wave][rowoff + r][16 + bcol] = (__bf16)p1;
        }
        v16bf pf = frag_split(prow + asel, prow + 16 + asel);

        // O += P (16x32) x Vt (32x16 per 16-wide d block)
#pragma unroll
        for (int t = 0; t < 4; ++t) {
            const __bf16* vp = Vb + (long)(t * 16 + bcol) * SEQ + m0 + bk;
            accO[t] = wmma_bf16(pf, frag_contig(vp), accO[t]);
        }
    }

    // Single end-of-loop row-sum reduction (lanes within each 16-lane half
    // hold partial sums of the same row set), then normalize and store.
    const int rbase = (lane >> 4) * 8;
#pragma unroll
    for (int r = 0; r < 8; ++r) {
        float l = lsum[r];
        l += __shfl_xor(l, 1, 32);
        l += __shfl_xor(l, 2, 32);
        l += __shfl_xor(l, 4, 32);
        l += __shfl_xor(l, 8, 32);
        float inv = 1.0f / l;
#pragma unroll
        for (int t = 0; t < 4; ++t) {
            int n = n0 + rbase + r;
            int c = h * HD + t * 16 + bcol;
            Oa[((long)b * SEQ + n) * DIM + c] = (__bf16)(accO[t][r] * inv);
        }
    }
}

// ---------------------------------------------------------------------------
// Projection + bias + residual: y[b,c,n] = x[b,c,n] + sum_k Oa[b,n,k]*wp[c,k] + bp[c]
// Block = 4 waves; wave handles 16 rows x 128 cols.
// ---------------------------------------------------------------------------
__global__ void proj_gemm(const __bf16* __restrict__ Oa, const __bf16* __restrict__ wp,
                          const float* __restrict__ bp, const float* __restrict__ x,
                          float* __restrict__ y) {
    const int lane = threadIdx.x & 31;
    const int wave = threadIdx.x >> 5;
    int bid = blockIdx.x;
    int n0 = (bid % (SEQ / 16)) * 16;
    int b  = bid / (SEQ / 16);
    int m0 = wave * 128;

    const int arow = lane & 15;
    const int asel = (lane >> 4) * 8;
    const int bcol = lane & 15;
    const int bk   = (lane >> 4) * 8;
    const __bf16* aptr = Oa + ((long)b * SEQ + n0 + arow) * DIM;

    v8f acc[8] = {};
    for (int k0 = 0; k0 < DIM; k0 += 32) {
        v16bf af = frag_split(aptr + k0 + asel, aptr + k0 + 16 + asel);
#pragma unroll
        for (int t = 0; t < 8; ++t) {
            const __bf16* bptr = wp + (long)(m0 + t * 16 + bcol) * DIM + k0 + bk;
            acc[t] = wmma_bf16(af, frag_contig(bptr), acc[t]);
        }
    }

    const int rbase = (lane >> 4) * 8;
#pragma unroll
    for (int t = 0; t < 8; ++t) {
        int c = m0 + t * 16 + bcol;
        float bias = bp[c];
#pragma unroll
        for (int r = 0; r < 8; ++r) {
            int n = n0 + rbase + r;
            long xi = ((long)b * DIM + c) * SEQ + n;
            y[xi] = x[xi] + acc[t][r] + bias;
        }
    }
}

// ---------------------------------------------------------------------------
extern "C" void kernel_launch(void* const* d_in, const int* in_sizes, int n_in,
                              void* d_out, int out_size, void* d_ws, size_t ws_size,
                              hipStream_t stream) {
    const float* x     = (const float*)d_in[0];
    const float* wqkv  = (const float*)d_in[1];
    const float* bqkv  = (const float*)d_in[2];
    const float* wproj = (const float*)d_in[3];
    const float* bproj = (const float*)d_in[4];
    float* y = (float*)d_out;

    // Workspace carve (~86 MB total)
    char* ws = (char*)d_ws;
    size_t off = 0;
    auto carve = [&](size_t bytes) {
        char* p = ws + off;
        off += (bytes + 255) & ~(size_t)255;
        return p;
    };
    __bf16* xT  = (__bf16*)carve((size_t)BATCH * SEQ * DIM * 2);
    __bf16* wqb = (__bf16*)carve((size_t)3 * DIM * DIM * 2);
    __bf16* wpb = (__bf16*)carve((size_t)DIM * DIM * 2);
    __bf16* Q   = (__bf16*)carve((size_t)BATCH * HEADS * SEQ * HD * 2);
    __bf16* K   = (__bf16*)carve((size_t)BATCH * HEADS * SEQ * HD * 2);
    __bf16* Vt  = (__bf16*)carve((size_t)BATCH * HEADS * SEQ * HD * 2);
    __bf16* Oa  = (__bf16*)carve((size_t)BATCH * SEQ * DIM * 2);

    convert_f32_bf16<<<(3 * DIM * DIM + 255) / 256, 256, 0, stream>>>(wqkv, wqb, 3 * DIM * DIM);
    convert_f32_bf16<<<(DIM * DIM + 255) / 256, 256, 0, stream>>>(wproj, wpb, DIM * DIM);
    transpose_x<<<(BATCH * SEQ * DIM + 255) / 256, 256, 0, stream>>>(x, xT);

    qkv_gemm<<<BATCH * (SEQ / 16) * ((3 * DIM) / 256), 128, 0, stream>>>(xT, wqb, bqkv, Q, K, Vt);
    flash_attn<<<BATCH * HEADS * (SEQ / 64), 128, 0, stream>>>(Q, K, Vt, Oa);
    proj_gemm<<<BATCH * (SEQ / 16), 128, 0, stream>>>(Oa, wpb, bproj, x, y);
}
